// QuantumInspiredAttention_21612275433846
// MI455X (gfx1250) — matrix operand order repaired and checked by reading, not compiled, not run on hardware
//
#include <hip/hip_runtime.h>

// ---------------------------------------------------------------------------
// QuantumInspiredAttention on MI455X (gfx1250, wave32, WMMA + TDM).
//
// Identity: reference returns real( (attn + i*noise) @ V @ Wproj^T + b ).
// Wproj/bproj are real => the imaginary (noise) path never reaches the output.
// So this is exactly:  proj( softmax(Q K^T / sqrt(D)) @ V ).  The 256MB noise
// tensor is never read.
//
// bf16 WMMA (v_wmma_f32_16x16x32_bf16) with fp32 accumulation, fp32 softmax.
// Compute-bound (~172 GFLOP vs ~6us of HBM traffic at 23.3 TB/s), so bf16
// matrix cores are the right operating point. Weight/Key tiles are DMA'd into
// LDS by the Tensor Data Mover (TENSOR_LOAD_TO_LDS, TENSORcnt-tracked,
// double-buffered) so wave issue slots stay free for WMMA.
// ---------------------------------------------------------------------------

typedef __attribute__((ext_vector_type(16))) __bf16 v16bf;
typedef __attribute__((ext_vector_type(8)))  float  v8f;
typedef __attribute__((ext_vector_type(4)))  unsigned int v4u;
typedef __attribute__((ext_vector_type(8)))  int    v8i;
typedef __attribute__((ext_vector_type(4)))  int    v4i;

#if __has_builtin(__builtin_amdgcn_tensor_load_to_lds) && \
    __has_builtin(__builtin_amdgcn_s_wait_tensorcnt)
#define QIA_HAS_TDM 1
#else
#define QIA_HAS_TDM 0
#endif

static __device__ __forceinline__ unsigned short f2bf(float f) {
  // round-to-nearest-even fp32 -> bf16
  unsigned int u = __float_as_uint(f);
  unsigned int r = (u + 0x7FFFu + ((u >> 16) & 1u)) >> 16;
  return (unsigned short)r;
}

static __device__ __forceinline__ v16bf ld_frag(const unsigned short* p0,
                                                const unsigned short* p1) {
  union { uint4 u[2]; v16bf v; } t;
  t.u[0] = *(const uint4*)p0;
  t.u[1] = *(const uint4*)p1;
  return t.v;
}

static __device__ __forceinline__ v8f wmma_bf16(v16bf a, v16bf b, v8f c) {
  return __builtin_amdgcn_wmma_f32_16x16x32_bf16(
      /*neg_a=*/false, a, /*neg_b=*/false, b,
      /*c_mod=*/(short)0, c, /*reuse_a=*/false, /*reuse_b=*/false);
}

static __device__ __forceinline__ v8f v8f_zero() {
  v8f z;
#pragma unroll
  for (int i = 0; i < 8; ++i) z[i] = 0.0f;
  return z;
}

#if QIA_HAS_TDM
// Issue a 2D TENSOR_LOAD_TO_LDS: tile [tile_d1 rows x tile_d0 elems] of 2-byte
// elements, row stride stride0 (elems), from gaddr into LDS at lds_off.
// D# layout per CDNA5 ISA.
// This toolchain's builtin is the 6-arg form:
//   (v4u group0, v8i group1, v4i group2, v4i group3, v8i extra, i32 cpol)
static __device__ __forceinline__ void tdm_load_2d(unsigned lds_off,
                                                   unsigned long long gaddr,
                                                   unsigned tensor_d0,
                                                   unsigned tensor_d1,
                                                   unsigned tile_d0,
                                                   unsigned tile_d1,
                                                   unsigned long long stride0) {
  v4u g0;
  v8i g1;
  v4i g2 = (v4i)0;
  v4i g3 = (v4i)0;
  v8i g4 = (v8i)0;
  g0[0] = 1u;                                           // count=1, user mode
  g0[1] = lds_off;                                      // lds_addr (bytes)
  g0[2] = (unsigned)(gaddr & 0xffffffffull);            // global_addr[31:0]
  g0[3] = (unsigned)((gaddr >> 32) & 0x1ffffffull) | (2u << 30);  // type=2
  g1[0] = (int)(1u << 16);                              // data_size=1 (2B)
  g1[1] = (int)((tensor_d0 & 0xffffu) << 16);           // tensor_dim0 lo16
  g1[2] = (int)(((tensor_d0 >> 16) & 0xffffu) | ((tensor_d1 & 0xffffu) << 16));
  g1[3] = (int)(((tensor_d1 >> 16) & 0xffffu) | ((tile_d0 & 0xffffu) << 16));
  g1[4] = (int)(tile_d1 & 0xffffu);                     // tile_dim2=0
  g1[5] = (int)(stride0 & 0xffffffffull);               // dim0 stride lo32
  g1[6] = (int)((stride0 >> 32) & 0xffffull);           // stride hi16, stride1=0
  g1[7] = 0;
  __builtin_amdgcn_tensor_load_to_lds(g0, g1, g2, g3, g4, /*cpol=*/0);
}
#endif

// ---------------------------------------------------------------------------
// fp32 -> bf16 elementwise convert
// ---------------------------------------------------------------------------
__global__ void qia_f32_to_bf16(const float* __restrict__ in,
                                unsigned short* __restrict__ out, int n) {
  int i = blockIdx.x * blockDim.x + threadIdx.x;
  int stride = gridDim.x * blockDim.x;
  for (; i < n; i += stride) out[i] = f2bf(in[i]);
}

// ---------------------------------------------------------------------------
// C[M,N] = A[M,K](bf16) @ W[N,K](bf16)^T + bias; optional per-col scale for
// the Q columns (col < qcols -> *qscale), output bf16 or fp32.
// Block = 128x128 tile, 8 waves, K-step 32. W tile DMA'd to LDS by the TDM
// (double-buffered) when available, else cooperatively loaded.
// ---------------------------------------------------------------------------
__global__ void __launch_bounds__(256)
qia_gemm_bf16_nt(const unsigned short* __restrict__ A,
                 const unsigned short* __restrict__ W,
                 const float* __restrict__ bias,
                 void* __restrict__ Cout,
                 int M, int N, int K,
                 int out_bf16, int qcols, float qscale) {
  __shared__ __align__(16) unsigned short Ws[2][128 * 32];  // 2 x 8 KB

  const int tid  = threadIdx.x;
  const int lane = tid & 31;
  const int wave = tid >> 5;
  const int lh   = lane >> 4;   // which 16-lane half
  const int ln   = lane & 15;

  const int nb = blockIdx.x;    // N / 128
  const int mb = blockIdx.y;    // M / 128
  const int m0 = mb * 128 + wave * 16;
  const int nk = K >> 5;

  v8f acc[8];
#pragma unroll
  for (int j = 0; j < 8; ++j) acc[j] = v8f_zero();

  const unsigned short* Arow = A + (size_t)(m0 + ln) * K;

#if QIA_HAS_TDM
  const unsigned long long wbase =
      (unsigned long long)(uintptr_t)(W + (size_t)(nb * 128) * K);
  if (wave == 0) {
    tdm_load_2d((unsigned)(uintptr_t)&Ws[0][0], wbase,
                (unsigned)K, 128u, 32u, 128u, (unsigned long long)K);
  }
#endif

  for (int ki = 0; ki < nk; ++ki) {
#if QIA_HAS_TDM
    if (wave == 0) {
      if (ki + 1 < nk) {
        tdm_load_2d((unsigned)(uintptr_t)&Ws[(ki + 1) & 1][0],
                    wbase + (unsigned long long)(ki + 1) * 64ull,
                    (unsigned)K, 128u, 32u, 128u, (unsigned long long)K);
        __builtin_amdgcn_s_wait_tensorcnt(1);  // current tile complete
      } else {
        __builtin_amdgcn_s_wait_tensorcnt(0);
      }
    }
#else
    {
      int r   = tid >> 1;
      int seg = (tid & 1) * 16;
      const uint4* src =
          (const uint4*)(W + (size_t)(nb * 128 + r) * K + ki * 32 + seg);
      uint4 a0 = src[0];
      uint4 a1 = src[1];
      uint4* dst = (uint4*)(&Ws[ki & 1][0] + r * 32 + seg);
      dst[0] = a0;
      dst[1] = a1;
    }
#endif
    __syncthreads();

    const unsigned short* Wcur = &Ws[ki & 1][0];

    // A fragment: 16x32, lane ln = row, halves interleaved (k: +0..7, +16..23)
    const int kbase = ki * 32 + lh * 8;
    v16bf af = ld_frag(Arow + kbase, Arow + kbase + 16);

    // hoist all 8 B fragments so DS loads pipeline ahead of the WMMAs
    v16bf bfr[8];
#pragma unroll
    for (int jt = 0; jt < 8; ++jt) {
      const unsigned short* wp = Wcur + (jt * 16 + ln) * 32 + lh * 8;
      bfr[jt] = ld_frag(wp, wp + 16);
    }
#pragma unroll
    for (int jt = 0; jt < 8; ++jt) acc[jt] = wmma_bf16(af, bfr[jt], acc[jt]);

    __syncthreads();
  }

  // epilogue: C layout => element (vgpr r, lane): row = m0 + r + 8*lh, col = ln
#pragma unroll
  for (int jt = 0; jt < 8; ++jt) {
    const int col = nb * 128 + jt * 16 + ln;
    const float bcol = bias ? bias[col] : 0.0f;
    const float s = (col < qcols) ? qscale : 1.0f;
#pragma unroll
    for (int r = 0; r < 8; ++r) {
      const int row = m0 + r + lh * 8;
      const float v = (acc[jt][r] + bcol) * s;
      if (out_bf16)
        ((unsigned short*)Cout)[(size_t)row * N + col] = f2bf(v);
      else
        ((float*)Cout)[(size_t)row * N + col] = v;
    }
  }
}

// ---------------------------------------------------------------------------
// Flash attention: qkv [4096, 6144] bf16 (Q pre-scaled by 1/sqrt(D)).
// Grid: (8 row-blocks, B*H). Block: 256 threads (8 waves x 16 query rows).
// K block DMA'd by TDM (overlapped with the manual V transpose), online
// softmax, all matmuls via v_wmma_f32_16x16x32_bf16.
// Output: attn_out [4096, 2048] bf16 in [B,N,H,D] layout.
// ---------------------------------------------------------------------------
__global__ void __launch_bounds__(256)
qia_flash_attn(const unsigned short* __restrict__ qkv,
               unsigned short* __restrict__ out) {
  const int N = 1024, D = 128, C3 = 6144, C = 2048;

  __shared__ __align__(16) unsigned short Ks[64 * 128];    // K block, row-major
  __shared__ __align__(16) unsigned short Vt[128 * 64];    // V transposed [d][j]
  __shared__ __align__(16) unsigned short Pw[8 * 16 * 64]; // per-wave P tiles

  const int tid  = threadIdx.x;
  const int lane = tid & 31;
  const int wave = tid >> 5;
  const int lh   = lane >> 4;
  const int ln   = lane & 15;

  const int rb = blockIdx.x;        // query row block (0..7)
  const int b  = blockIdx.y >> 4;
  const int h  = blockIdx.y & 15;

  // Load Q fragments for this wave's 16 rows (kept in registers, 4 k-steps)
  const int qrow = rb * 128 + wave * 16 + ln;
  const unsigned short* qp = qkv + (size_t)(b * N + qrow) * C3 + h * D;
  v16bf qf[4];
#pragma unroll
  for (int t = 0; t < 4; ++t) {
    const int kb = t * 32 + lh * 8;
    qf[t] = ld_frag(qp + kb, qp + kb + 16);
  }

  v8f O[8];
#pragma unroll
  for (int dt = 0; dt < 8; ++dt) O[dt] = v8f_zero();
  float mi[8], li[8];
#pragma unroll
  for (int r = 0; r < 8; ++r) { mi[r] = -1e30f; li[r] = 0.0f; }

  unsigned short* Pmine = Pw + wave * 16 * 64;

  for (int jb = 0; jb < 16; ++jb) {
    const int j0 = jb * 64;

    // ---- stage K (TDM DMA) and V-transposed (manual) into LDS ----
#if QIA_HAS_TDM
    if (wave == 0) {
      const unsigned long long kaddr =
          (unsigned long long)(uintptr_t)(qkv + (size_t)(b * N + j0) * C3 + C +
                                          h * D);
      tdm_load_2d((unsigned)(uintptr_t)&Ks[0], kaddr,
                  (unsigned)C3, 64u, 128u, 64u, (unsigned long long)C3);
    }
#endif
    {
      const int r   = tid >> 2;          // 0..63 key row
      const int seg = (tid & 3) * 32;    // 0,32,64,96 (elements)
#if !QIA_HAS_TDM
      const uint4* ksrc =
          (const uint4*)(qkv + (size_t)(b * N + j0 + r) * C3 + C + h * D + seg);
      uint4* kdst = (uint4*)(Ks + r * 128 + seg);
      kdst[0] = ksrc[0]; kdst[1] = ksrc[1]; kdst[2] = ksrc[2]; kdst[3] = ksrc[3];
#endif
      union { uint4 u[4]; unsigned short e[32]; } vv;
      const uint4* vsrc =
          (const uint4*)(qkv + (size_t)(b * N + j0 + r) * C3 + 2 * C + h * D + seg);
      vv.u[0] = vsrc[0]; vv.u[1] = vsrc[1]; vv.u[2] = vsrc[2]; vv.u[3] = vsrc[3];
#pragma unroll
      for (int d2 = 0; d2 < 32; ++d2) Vt[(seg + d2) * 64 + r] = vv.e[d2];
    }
#if QIA_HAS_TDM
    if (wave == 0) __builtin_amdgcn_s_wait_tensorcnt(0);
#endif
    __syncthreads();

    // S = Q K^T : 4 16x16 subtiles over 64 keys, K-dim = D = 128 (4 steps)
    v8f S[4];
#pragma unroll
    for (int jt = 0; jt < 4; ++jt) {
      v8f s = v8f_zero();
#pragma unroll
      for (int t = 0; t < 4; ++t) {
        const unsigned short* kp = Ks + (jt * 16 + ln) * 128 + t * 32 + lh * 8;
        v16bf bfK = ld_frag(kp, kp + 16);
        s = wmma_bf16(qf[t], bfK, s);
      }
      S[jt] = s;
    }

    // ---- online softmax ----
    float mb[8];
#pragma unroll
    for (int r = 0; r < 8; ++r)
      mb[r] = fmaxf(fmaxf(S[0][r], S[1][r]), fmaxf(S[2][r], S[3][r]));
#pragma unroll
    for (int mask = 1; mask < 16; mask <<= 1)
#pragma unroll
      for (int r = 0; r < 8; ++r)
        mb[r] = fmaxf(mb[r], __shfl_xor(mb[r], mask, 32));

    float corr[8];
#pragma unroll
    for (int r = 0; r < 8; ++r) {
      const float mn = fmaxf(mi[r], mb[r]);
      corr[r] = __expf(mi[r] - mn);
      mi[r] = mn;
    }

    float lb[8];
#pragma unroll
    for (int r = 0; r < 8; ++r) lb[r] = 0.0f;
#pragma unroll
    for (int jt = 0; jt < 4; ++jt) {
#pragma unroll
      for (int r = 0; r < 8; ++r) {
        const float p = __expf(S[jt][r] - mi[r]);
        lb[r] += p;
        Pmine[(r + lh * 8) * 64 + jt * 16 + ln] = f2bf(p);
      }
    }
#pragma unroll
    for (int mask = 1; mask < 16; mask <<= 1)
#pragma unroll
      for (int r = 0; r < 8; ++r) lb[r] += __shfl_xor(lb[r], mask, 32);
#pragma unroll
    for (int r = 0; r < 8; ++r) li[r] = li[r] * corr[r] + lb[r];
#pragma unroll
    for (int dt = 0; dt < 8; ++dt)
#pragma unroll
      for (int r = 0; r < 8; ++r) O[dt][r] *= corr[r];

    // ---- O += P V : A = P (via LDS, same-wave LDS ops are in-order) ----
#pragma unroll
    for (int dt = 0; dt < 8; ++dt) {
#pragma unroll
      for (int t = 0; t < 2; ++t) {
        const unsigned short* pp = Pmine + ln * 64 + t * 32 + lh * 8;
        v16bf afP = ld_frag(pp, pp + 16);
        const unsigned short* vp = Vt + (dt * 16 + ln) * 64 + t * 32 + lh * 8;
        v16bf bfV = ld_frag(vp, vp + 16);
        O[dt] = wmma_bf16(afP, bfV, O[dt]);
      }
    }
    __syncthreads();  // protect Ks/Vt before next iteration's loads
  }

  // normalize and store: out[b, n, h, d] as bf16
  float inv[8];
#pragma unroll
  for (int r = 0; r < 8; ++r) inv[r] = 1.0f / li[r];
#pragma unroll
  for (int dt = 0; dt < 8; ++dt) {
    const int d = dt * 16 + ln;
#pragma unroll
    for (int r = 0; r < 8; ++r) {
      const int row = rb * 128 + wave * 16 + r + lh * 8;
      out[(size_t)(b * N + row) * C + h * D + d] = f2bf(O[dt][r] * inv[r]);
    }
  }
}

// ---------------------------------------------------------------------------
// Launch
// ---------------------------------------------------------------------------
extern "C" void kernel_launch(void* const* d_in, const int* in_sizes, int n_in,
                              void* d_out, int out_size, void* d_ws, size_t ws_size,
                              hipStream_t stream) {
  const float* x     = (const float*)d_in[0];  // [4, 1024, 2048]
  const float* Wqkv  = (const float*)d_in[1];  // [6144, 2048]
  const float* bqkv  = (const float*)d_in[2];  // [6144]
  const float* Wproj = (const float*)d_in[3];  // [2048, 2048]
  const float* bproj = (const float*)d_in[4];  // [2048]
  // d_in[5] = noise: provably dead (imaginary part discarded by .real)

  const int M = 4096;      // B*N
  const int C = 2048;
  const int C3 = 6144;

  char* ws = (char*)d_ws;
  unsigned short* xb     = (unsigned short*)(ws);                 // 16 MB
  unsigned short* wqkvb  = (unsigned short*)(ws + 16777216);      // 24 MB
  unsigned short* wprojb = (unsigned short*)(ws + 41943040);      //  8 MB
  unsigned short* qkvb   = (unsigned short*)(ws + 50331648);      // 48 MB
  unsigned short* attnb  = (unsigned short*)(ws + 100663296);     // 16 MB

  // fp32 -> bf16 conversions
  qia_f32_to_bf16<<<4096, 256, 0, stream>>>(x, xb, M * C);
  qia_f32_to_bf16<<<4096, 256, 0, stream>>>(Wqkv, wqkvb, C3 * C);
  qia_f32_to_bf16<<<2048, 256, 0, stream>>>(Wproj, wprojb, C * C);

  // qkv = x @ Wqkv^T + bqkv  (Q columns pre-scaled by 1/sqrt(128))
  qia_gemm_bf16_nt<<<dim3(C3 / 128, M / 128), 256, 0, stream>>>(
      xb, wqkvb, bqkv, (void*)qkvb, M, C3, C,
      /*out_bf16=*/1, /*qcols=*/2048, /*qscale=*/0.08838834764831845f);

  // attention (flash, online softmax)
  qia_flash_attn<<<dim3(8, 64), 256, 0, stream>>>(qkvb, attnb);

  // out = attn_out @ Wproj^T + bproj  (fp32 output)
  qia_gemm_bf16_nt<<<dim3(C / 128, M / 128), 256, 0, stream>>>(
      attnb, wprojb, bproj, d_out, M, C, C,
      /*out_bf16=*/0, /*qcols=*/0, /*qscale=*/1.0f);
}
